// MoELayer_32091995635896
// MI455X (gfx1250) — compile-verified, matrix operand order checked by reading, not run on hardware
//
#include <hip/hip_runtime.h>
#include <hip/hip_bf16.h>

// ---------------------------------------------------------------------------
// MoE layer for MI455X (gfx1250), wave32 + WMMA bf16 + async LDS copies.
//   NTOK=8192 tokens, D=I=1024, 14 routed experts (top-2), 2 shared experts.
// Pipeline:
//   1. zero per-expert counters
//   2. out = x (residual) and xb = bf16(x)
//   3. router: logits -> softmax -> top2 -> gates + gathered token lists
//   4. per expert pass (2 shared + 14 routed):
//        MODE1: H = gelu(gather(X) @ W1 + b1)        (bf16 out)
//        MODE2: out += gate * (H @ W2 + b2)          (plain RMW, launches
//                                                     serialize on stream)
// GEMMs: 128x128x32 tiles, double-buffered LDS, v_wmma_f32_16x16x32_bf16,
// A tile staged with global_load_async_to_lds_b128 when available.
// ---------------------------------------------------------------------------

#define NTOK   8192
#define DDIM   1024
#define IDIM   1024
#define NEXP   14
#define NSH    2

#define BM     128
#define BN     128
#define BK     32
#define KDIM   1024          // K of both GEMMs (D and I are both 1024)
#define NCOLS  1024          // N of both GEMMs
#define KSTEPS (KDIM / BK)
#define APAD   8

typedef __bf16 bf16_t;
typedef bf16_t v16bf __attribute__((ext_vector_type(16)));
typedef float  v8f   __attribute__((ext_vector_type(8)));
typedef int    v4i_t __attribute__((ext_vector_type(4)));

union FragAB {
    uint4 q[2];
    v16bf v;
};

#if defined(__HIP_DEVICE_COMPILE__) && \
    __has_builtin(__builtin_amdgcn_global_load_async_to_lds_b128)
#define HAS_ASYNC_LDS 1
#else
#define HAS_ASYNC_LDS 0
#endif

#if HAS_ASYNC_LDS
typedef __attribute__((address_space(1))) v4i_t* gptr_v4i;
typedef __attribute__((address_space(3))) v4i_t* lptr_v4i;

__device__ __forceinline__ gptr_v4i as_global_v4i(const void* p) {
    return (gptr_v4i)(unsigned long long)(uintptr_t)p;
}
__device__ __forceinline__ lptr_v4i as_lds_v4i(void* p) {
    // generic LDS pointer: addr[31:0] is the LDS offset (ISA 10.2 aperture map)
    return (lptr_v4i)(unsigned int)(uintptr_t)p;
}
__device__ __forceinline__ void wait_async0() {
#if __has_builtin(__builtin_amdgcn_s_wait_asynccnt)
    __builtin_amdgcn_s_wait_asynccnt(0);
#else
    asm volatile("s_wait_asynccnt 0" ::: "memory");
#endif
}
#endif

__device__ __forceinline__ unsigned short f2bf(float f) {
    unsigned int u = __float_as_uint(f);
    unsigned int r = u + 0x7FFFu + ((u >> 16) & 1u);   // round-to-nearest-even
    return (unsigned short)(r >> 16);
}

__device__ __forceinline__ float gelu_exact(float v) {
    return 0.5f * v * (1.0f + erff(v * 0.70710678118654752f));
}

// ---------------------------------------------------------------------------
// Small kernels
// ---------------------------------------------------------------------------

__global__ void zero_counts_kernel(int* __restrict__ counts) {
    if (threadIdx.x < 16) counts[threadIdx.x] = 0;
}

// out = x (residual term), xb = bf16(x)
__global__ void init_cast_kernel(const float* __restrict__ x,
                                 float* __restrict__ out,
                                 unsigned short* __restrict__ xb, int n) {
    for (int i = blockIdx.x * blockDim.x + threadIdx.x; i < n;
         i += gridDim.x * blockDim.x) {
        float v = x[i];
        out[i] = v;
        xb[i]  = f2bf(v);
    }
}

// One wave32 per token: router logits, softmax, top-2, gather-list append.
__global__ void router_kernel(const float* __restrict__ x,
                              const float* __restrict__ rw,
                              const float* __restrict__ rb,
                              float* __restrict__ gates,
                              int* __restrict__ counts,
                              int* __restrict__ tok_list) {
    const int wid   = threadIdx.x >> 5;
    const int lane  = threadIdx.x & 31;
    const int token = blockIdx.x * (blockDim.x >> 5) + wid;
    if (token >= NTOK) return;

    float acc[NEXP];
#pragma unroll
    for (int e = 0; e < NEXP; ++e) acc[e] = 0.0f;

    const float* xr = x + (size_t)token * DDIM;
    for (int d = lane; d < DDIM; d += 32) {
        float xv = xr[d];
        const float* w = rw + (size_t)d * NEXP;
#pragma unroll
        for (int e = 0; e < NEXP; ++e) acc[e] += xv * w[e];
    }
#pragma unroll
    for (int e = 0; e < NEXP; ++e) {
#pragma unroll
        for (int off = 16; off > 0; off >>= 1)
            acc[e] += __shfl_xor(acc[e], off, 32);
    }

    if (lane == 0) {
        float lg[NEXP];
        float mx = -3.0e38f;
#pragma unroll
        for (int e = 0; e < NEXP; ++e) {
            lg[e] = acc[e] + rb[e];
            mx = fmaxf(mx, lg[e]);
        }
        float s = 0.0f;
#pragma unroll
        for (int e = 0; e < NEXP; ++e) {
            lg[e] = __expf(lg[e] - mx);
            s += lg[e];
        }
        float inv = 1.0f / s;
        int i0 = 0;
#pragma unroll
        for (int e = 1; e < NEXP; ++e)
            if (lg[e] > lg[i0]) i0 = e;
        int i1 = (i0 == 0) ? 1 : 0;
#pragma unroll
        for (int e = 0; e < NEXP; ++e)
            if (e != i0 && lg[e] > lg[i1]) i1 = e;

        int p0 = atomicAdd(&counts[i0], 1);
        tok_list[(size_t)i0 * NTOK + p0] = token;
        gates[(size_t)i0 * NTOK + token] = lg[i0] * inv;

        int p1 = atomicAdd(&counts[i1], 1);
        tok_list[(size_t)i1 * NTOK + p1] = token;
        gates[(size_t)i1 * NTOK + token] = lg[i1] * inv;
    }
}

// ---------------------------------------------------------------------------
// Fused expert GEMM.
//  MODE 1: H[j,:] = gelu( A[tok(j),:] @ W + b )        (A = bf16 X, gathered)
//  MODE 2: out[tok(j),:] += gate * ( A[j,:] @ W + b )  (A = bf16 H, direct)
//  A: bf16 row-major [*, KDIM].  W: fp32 row-major [KDIM, NCOLS] (bf16'd in
//  LDS stage).  Block tile 128x128, BK=32 == one WMMA K.  8 waves: 4(M)x2(N),
//  each wave 32x64 = 8 v_wmma_f32_16x16x32_bf16 per K step.
// ---------------------------------------------------------------------------
template <int MODE>
__global__ void __launch_bounds__(256)
ffn_gemm_kernel(const unsigned short* __restrict__ Abf,
                const float* __restrict__ W,
                const float* __restrict__ bias,
                const int* __restrict__ tok_list,
                const int* __restrict__ count_ptr,
                int max_rows,
                const float* __restrict__ gates,
                unsigned short* __restrict__ Hout,
                float* __restrict__ out) {
    const int count = count_ptr ? *count_ptr : max_rows;
    const int blk_m = blockIdx.x;
    const int blk_n = blockIdx.y;
    if (blk_m * BM >= count) return;   // gathered pass: skip empty tiles

    __shared__ unsigned short As[2][BM][BK + APAD];   // [row][k]
    __shared__ unsigned short Bs[2][BN][BK + APAD];   // [col][k] (K-innermost)

    const int tid  = threadIdx.x;
    const int wid  = tid >> 5;
    const int lane = tid & 31;
    const int wm0  = (wid & 3) * 32;   // wave M origin in tile
    const int wn0  = (wid >> 2) * 64;  // wave N origin in tile
    const int lr   = lane & 15;
    const int hi   = lane >> 4;
    const int kb   = hi * 8;           // per-lane K sub-offset (ISA A/B layout)

    // A-tile addressing: 2 threads/row, 32 B each (one row = 32 bf16 = 64 B).
    const int a_row  = tid >> 1;
    const int a_half = tid & 1;
    const int a_j    = blk_m * BM + a_row;
    int a_src_row = a_j;
    if (MODE == 1 && tok_list && a_j < count) a_src_row = tok_list[a_j];
    const unsigned short* a_src_base =
        Abf + (size_t)a_src_row * KDIM + a_half * 16;

    // Stage A tile for K-offset k0 into buffer `buf`.
    auto stage_a = [&](int buf, int k0) {
        unsigned short* dst = &As[buf][a_row][a_half * 16];
        if (a_j < count) {
#if HAS_ASYNC_LDS
            // 32 B per thread: two async b128, same imm offset applied to
            // both the global and LDS address (ISA 10.7 async addressing).
            __builtin_amdgcn_global_load_async_to_lds_b128(
                as_global_v4i(a_src_base + k0), as_lds_v4i(dst), 0, 0);
            __builtin_amdgcn_global_load_async_to_lds_b128(
                as_global_v4i(a_src_base + k0), as_lds_v4i(dst), 16, 0);
#else
            const uint4* src = (const uint4*)(a_src_base + k0);
            uint4* d4 = (uint4*)dst;
            d4[0] = src[0];
            d4[1] = src[1];
#endif
        }
        // rows >= count: leave LDS stale; those C rows are never stored.
    };

    // Stage B tile (fp32 -> bf16, transposed to [col][k], k-pairs packed
    // into dword stores): thread handles 2 k-rows x 8 cols.
    const int b_kp  = tid >> 4;        // 0..15 -> k rows 2*kp, 2*kp+1
    const int b_seg = tid & 15;        // 0..15 -> cols seg*8 .. seg*8+7
    auto stage_b = [&](int buf, int k0) {
        const float* s0 = W + (size_t)(k0 + 2 * b_kp) * NCOLS +
                          blk_n * BN + b_seg * 8;
        const float* s1 = s0 + NCOLS;
        float4 e0 = ((const float4*)s0)[0];
        float4 e1 = ((const float4*)s0)[1];
        float4 o0 = ((const float4*)s1)[0];
        float4 o1 = ((const float4*)s1)[1];
        unsigned int pk[8];
        pk[0] = f2bf(e0.x) | ((unsigned int)f2bf(o0.x) << 16);
        pk[1] = f2bf(e0.y) | ((unsigned int)f2bf(o0.y) << 16);
        pk[2] = f2bf(e0.z) | ((unsigned int)f2bf(o0.z) << 16);
        pk[3] = f2bf(e0.w) | ((unsigned int)f2bf(o0.w) << 16);
        pk[4] = f2bf(e1.x) | ((unsigned int)f2bf(o1.x) << 16);
        pk[5] = f2bf(e1.y) | ((unsigned int)f2bf(o1.y) << 16);
        pk[6] = f2bf(e1.z) | ((unsigned int)f2bf(o1.z) << 16);
        pk[7] = f2bf(e1.w) | ((unsigned int)f2bf(o1.w) << 16);
#pragma unroll
        for (int c = 0; c < 8; ++c)
            *(unsigned int*)&Bs[buf][b_seg * 8 + c][2 * b_kp] = pk[c];
        // warm L2 for the tile two K-steps out (global_prefetch_b8)
        __builtin_prefetch((const void*)(s0 + 2 * (size_t)BK * NCOLS), 0, 1);
    };

    v8f c[2][4];
#pragma unroll
    for (int mf = 0; mf < 2; ++mf)
#pragma unroll
        for (int nf = 0; nf < 4; ++nf)
            c[mf][nf] = (v8f)(0.0f);

    stage_a(0, 0);
    stage_b(0, 0);
#if HAS_ASYNC_LDS
    wait_async0();
#endif
    __syncthreads();

    for (int kk = 0; kk < KSTEPS; ++kk) {
        const int cur = kk & 1;
        if (kk + 1 < KSTEPS) {
            stage_a(cur ^ 1, (kk + 1) * BK);
            stage_b(cur ^ 1, (kk + 1) * BK);
        }

        FragAB a[2], b[4];
#pragma unroll
        for (int mf = 0; mf < 2; ++mf) {
            const unsigned short* p = &As[cur][wm0 + mf * 16 + lr][kb];
            a[mf].q[0] = *(const uint4*)p;
            a[mf].q[1] = *(const uint4*)(p + 16);
        }
#pragma unroll
        for (int nf = 0; nf < 4; ++nf) {
            const unsigned short* p = &Bs[cur][wn0 + nf * 16 + lr][kb];
            b[nf].q[0] = *(const uint4*)p;
            b[nf].q[1] = *(const uint4*)(p + 16);
        }
#pragma unroll
        for (int mf = 0; mf < 2; ++mf)
#pragma unroll
            for (int nf = 0; nf < 4; ++nf)
                c[mf][nf] = __builtin_amdgcn_wmma_f32_16x16x32_bf16(
                    false, a[mf].v, false, b[nf].v, (short)0, c[mf][nf],
                    false, false);

#if HAS_ASYNC_LDS
        wait_async0();   // next-buffer async writes landed before publish
#endif
        __syncthreads();
    }

    // --- epilogue ------------------------------------------------------
    float bv[4];
#pragma unroll
    for (int nf = 0; nf < 4; ++nf)
        bv[nf] = bias[blk_n * BN + wn0 + nf * 16 + lr];

#pragma unroll
    for (int mf = 0; mf < 2; ++mf) {
#pragma unroll
        for (int r = 0; r < 8; ++r) {
            // C layout: vgpr r -> M = r + 8*hi, N = lane&15 (ISA 7.12.2)
            int jj = blk_m * BM + wm0 + mf * 16 + r + 8 * hi;
            if (jj >= count) continue;
            if (MODE == 1) {
#pragma unroll
                for (int nf = 0; nf < 4; ++nf) {
                    int col = blk_n * BN + wn0 + nf * 16 + lr;
                    float v = gelu_exact(c[mf][nf][r] + bv[nf]);
                    Hout[(size_t)jj * NCOLS + col] = f2bf(v);
                }
            } else {
                int tok = tok_list ? tok_list[jj] : jj;
                float g = gates ? gates[tok] : 1.0f;
#pragma unroll
                for (int nf = 0; nf < 4; ++nf) {
                    int col = blk_n * BN + wn0 + nf * 16 + lr;
                    // one writer per (tok,col) per launch; launches are
                    // stream-ordered -> plain RMW, no atomics needed
                    float* o = out + (size_t)tok * NCOLS + col;
                    *o = *o + g * (c[mf][nf][r] + bv[nf]);
                }
            }
        }
    }
}

// ---------------------------------------------------------------------------
// Host launcher
// ---------------------------------------------------------------------------
extern "C" void kernel_launch(void* const* d_in, const int* in_sizes, int n_in,
                              void* d_out, int out_size, void* d_ws,
                              size_t ws_size, hipStream_t stream) {
    const float* x         = (const float*)d_in[0];
    const float* router_w  = (const float*)d_in[1];
    const float* router_b  = (const float*)d_in[2];
    const float* shared_w1 = (const float*)d_in[3];
    const float* shared_b1 = (const float*)d_in[4];
    const float* shared_w2 = (const float*)d_in[5];
    const float* shared_b2 = (const float*)d_in[6];
    const float* routed_w1 = (const float*)d_in[7];
    const float* routed_b1 = (const float*)d_in[8];
    const float* routed_w2 = (const float*)d_in[9];
    const float* routed_b2 = (const float*)d_in[10];
    float* out = (float*)d_out;

    // workspace carve-up (~35 MB)
    char* p = (char*)d_ws;
    unsigned short* xb = (unsigned short*)p; p += (size_t)NTOK * DDIM * 2;
    unsigned short* hb = (unsigned short*)p; p += (size_t)NTOK * IDIM * 2;
    float* gates = (float*)p;                p += (size_t)NEXP * NTOK * 4;
    int* toklist = (int*)p;                  p += (size_t)NEXP * NTOK * 4;
    int* counts  = (int*)p;                  p += 256;

    zero_counts_kernel<<<1, 32, 0, stream>>>(counts);
    init_cast_kernel<<<2048, 256, 0, stream>>>(x, out, xb, NTOK * DDIM);
    router_kernel<<<NTOK / 8, 256, 0, stream>>>(x, router_w, router_b, gates,
                                                counts, toklist);

    dim3 grid(NTOK / BM, NCOLS / BN);   // 64 x 8

    for (int s = 0; s < NSH; ++s) {
        ffn_gemm_kernel<1><<<grid, 256, 0, stream>>>(
            xb, shared_w1 + (size_t)s * DDIM * IDIM, shared_b1 + (size_t)s * IDIM,
            nullptr, nullptr, NTOK, nullptr, hb, nullptr);
        ffn_gemm_kernel<2><<<grid, 256, 0, stream>>>(
            hb, shared_w2 + (size_t)s * IDIM * DDIM, shared_b2 + (size_t)s * DDIM,
            nullptr, nullptr, NTOK, nullptr, nullptr, out);
    }
    for (int e = 0; e < NEXP; ++e) {
        ffn_gemm_kernel<1><<<grid, 256, 0, stream>>>(
            xb, routed_w1 + (size_t)e * DDIM * IDIM, routed_b1 + (size_t)e * IDIM,
            toklist + (size_t)e * NTOK, counts + e, NTOK, nullptr, hb, nullptr);
        ffn_gemm_kernel<2><<<grid, 256, 0, stream>>>(
            hb, routed_w2 + (size_t)e * IDIM * DDIM, routed_b2 + (size_t)e * DDIM,
            toklist + (size_t)e * NTOK, counts + e, NTOK,
            gates + (size_t)e * NTOK, nullptr, out);
    }
}
